// predict_84335977824863
// MI455X (gfx1250) — matrix operand order repaired and checked by reading, not compiled, not run on hardware
//
#include <hip/hip_runtime.h>
#include <hip/hip_bf16.h>

typedef __attribute__((ext_vector_type(16))) _Float16 v16h;
typedef __attribute__((ext_vector_type(8)))  float    v8f;

#define NEG_SLOPE 0.1f
#define TILE_W 64
#define CH_T   32
#define LDS_W  66   // TILE_W + 2 halo

static __device__ __forceinline__ float leaky_f(float r, int f) {
    return f ? ((r >= 0.0f) ? r : NEG_SLOPE * r) : r;
}

// ---------------------------------------------------------------------------
// Implicit-GEMM 3x3 SAME conv, f16 NHWC activations, WMMA f16->f32.
// Block = 128 threads (4 waves): 1 output row, 64-pixel strip, 16 out-channels.
// Input tile staged with GLOBAL_LOAD_ASYNC_TO_LDS_B128 (ASYNCcnt); halo lanes
// read a zeroed page. K = Cin*9 tiled as (32-ch chunk) x (9 taps).
// ---------------------------------------------------------------------------
__global__ __launch_bounds__(128)
void conv3x3_wmma_kernel(const _Float16* __restrict__ in,   // NHWC, stride Cins
                         const _Float16* __restrict__ wgt,  // [O][9][Cin]
                         const float* __restrict__ bias,
                         _Float16* __restrict__ out,        // NHWC, stride Couts
                         const _Float16* __restrict__ zpage,
                         int Cin, int Cins, int Cout, int Couts,
                         int H, int W, int leaky_flag)
{
    const int tilesW = (W + TILE_W - 1) / TILE_W;
    const int bx     = blockIdx.x % tilesW;
    const int co_t   = blockIdx.x / tilesW;
    const int y      = blockIdx.y;
    const int n      = blockIdx.z;
    const int x0     = bx * TILE_W;
    const int cout0  = co_t * 16;

    __shared__ __align__(128) _Float16 lds_in[3][LDS_W][CH_T];  // [row][x][chan]
    __shared__ __align__(16)  _Float16 lds_w[16][9][CH_T];      // [cout][tap][chan]

    const int tid    = threadIdx.x;
    const int lane   = tid & 31;
    const int wid    = tid >> 5;
    const int m_lane = lane & 15;
    const int halfid = lane >> 4;

    const size_t hw  = (size_t)H * W;
    const _Float16* inp = in + (size_t)n * hw * Cins;
    const unsigned lds_in_base = (unsigned)(size_t)(&lds_in[0][0][0]);

    v8f acc = {};

    for (int c0 = 0; c0 < Cin; c0 += CH_T) {
        // --- async stage of input tile: 3 rows x 66 px x 32 ch (64B per pos) ---
        for (int idx = tid; idx < 3 * LDS_W * 4; idx += 128) {
            int q   = idx & 3;          // 16B sub-chunk -> channels q*8..q*8+7
            int pos = idx >> 2;
            int xx  = pos % LDS_W;
            int r   = pos / LDS_W;
            int gy  = y + r - 1;
            int gx  = x0 + xx - 1;
            bool ok = (gy >= 0) && (gy < H) && (gx >= 0) && (gx < W);
            const _Float16* gp = ok
                ? (inp + ((size_t)gy * W + gx) * Cins + (c0 + q * 8))
                : zpage;
            unsigned laddr = lds_in_base +
                (unsigned)(((r * LDS_W + xx) * CH_T + q * 8) * sizeof(_Float16));
            unsigned long long ga = (unsigned long long)(size_t)gp;
            asm volatile("global_load_async_to_lds_b128 %0, %1, off"
                         :: "v"(laddr), "v"(ga) : "memory");
        }
        // --- stage weight tile (already f16, [O][9][Cin]) ---
        for (int idx = tid; idx < 16 * 9 * CH_T; idx += 128) {
            int o = idx / (9 * CH_T);
            int t = (idx / CH_T) % 9;
            int c = idx % CH_T;
            int go = cout0 + o, gc = c0 + c;
            _Float16 v = (_Float16)0.0f;
            if (go < Cout && gc < Cin)
                v = wgt[((size_t)go * 9 + t) * Cin + gc];
            lds_w[o][t][c] = v;
        }
        asm volatile("s_wait_asynccnt 0" ::: "memory");
        __syncthreads();

        const int px = wid * 16 + m_lane;

        #pragma unroll
        for (int tap = 0; tap < 9; ++tap) {
            const int dy = tap / 3, dx = tap % 3;
            v16h a, b;
            #pragma unroll
            for (int e = 0; e < 16; ++e) {
                int r = e >> 1, sub = e & 1;
                int kA = (halfid << 4) + e;
                a[e] = lds_in[dy][px + dx][kA];
                int kB = (halfid << 4) + e;
                b[e] = lds_w[m_lane][tap][kB];
            }
            acc = __builtin_amdgcn_wmma_f32_16x16x32_f16(
                      false, a, false, b, (short)0, acc, false, false);
        }
        __syncthreads();
    }

    // --- epilogue: bias, leaky, f16 NHWC store (lane n = cout -> coalesced) ---
    const int cout = cout0 + m_lane;
    if (cout < Cout) {
        const float bv = bias ? bias[cout] : 0.0f;
        _Float16* ob = out + ((size_t)n * hw + (size_t)y * W) * Couts + cout;
        #pragma unroll
        for (int v = 0; v < 8; ++v) {
            int m = v + (halfid << 3);
            int x = x0 + wid * 16 + m;
            if (x < W)
                ob[(size_t)x * Couts] = (_Float16)leaky_f(acc[v] + bv, leaky_flag);
        }
    }
}

// weight repack: f32 [O][I][3][3] -> f16 [O][9][I]
__global__ void wcvt_kernel(const float* __restrict__ src, _Float16* __restrict__ dst,
                            int O, int I)
{
    long total = (long)O * I * 9;
    long idx = (long)blockIdx.x * blockDim.x + threadIdx.x;
    if (idx >= total) return;
    int t = idx % 9;
    int c = (idx / 9) % I;
    int o = idx / ((long)9 * I);
    dst[((size_t)o * 9 + t) * I + c] = (_Float16)src[idx];
}

// pack f32 NCHW slab -> f16 NHWC (padded channel stride) at channel offset
__global__ void pack_nhwc_kernel(const float* __restrict__ src, _Float16* __restrict__ dst,
                                 int Csrc, int coff, int Cstride, int HW, int N)
{
    long total = (long)N * Csrc * HW;
    long idx = (long)blockIdx.x * blockDim.x + threadIdx.x;
    if (idx >= total) return;
    int p = idx % HW;
    int c = (idx / HW) % Csrc;
    int n = idx / ((long)HW * Csrc);
    dst[((size_t)n * HW + p) * Cstride + coff + c] = (_Float16)src[idx];
}

// ---------------------------------------------------------------------------
__global__ void up2x_scale_kernel(const float* __restrict__ in, float* __restrict__ out,
                                  int C, int h, int w, int N, float scale)
{
    int oh = 2 * h, ow = 2 * w;
    long total = (long)N * C * oh * ow;
    long idx = (long)blockIdx.x * blockDim.x + threadIdx.x;
    if (idx >= total) return;
    int ox = idx % ow;
    int oy = (idx / ow) % oh;
    int c  = (idx / ((long)ow * oh)) % C;
    int n  = idx / ((long)ow * oh * C);
    float sy = fmaxf((oy + 0.5f) * 0.5f - 0.5f, 0.0f);
    float sx = fmaxf((ox + 0.5f) * 0.5f - 0.5f, 0.0f);
    int y0 = (int)floorf(sy); int y1 = min(y0 + 1, h - 1);
    int x0 = (int)floorf(sx); int x1 = min(x0 + 1, w - 1);
    float wy = sy - (float)y0, wx = sx - (float)x0;
    const float* p = in + ((size_t)n * C + c) * h * w;
    float top = p[(size_t)y0 * w + x0] * (1.0f - wx) + p[(size_t)y0 * w + x1] * wx;
    float bot = p[(size_t)y1 * w + x0] * (1.0f - wx) + p[(size_t)y1 * w + x1] * wx;
    out[idx] = (top * (1.0f - wy) + bot * wy) * scale;
}

__global__ void splat_scatter_kernel(const float* __restrict__ val, const float* __restrict__ flow,
                                     float* __restrict__ accum, int C, int H, int W, int N, int foff)
{
    long total = (long)N * H * W;
    long idx = (long)blockIdx.x * blockDim.x + threadIdx.x;
    if (idx >= total) return;
    int x = idx % W;
    int y = (idx / W) % H;
    int n = idx / ((long)W * H);
    size_t hw = (size_t)H * W;
    const float* fl = flow + ((size_t)n * 4 + foff) * hw;
    float fx = fl[(size_t)y * W + x] + (float)x;
    float fy = fl[hw + (size_t)y * W + x] + (float)y;
    float x0f = floorf(fx), y0f = floorf(fy);
    float ax = fx - x0f, ay = fy - y0f;
    int xi[4], yi[4]; float wc[4];
    xi[0] = (int)x0f;     yi[0] = (int)y0f;     wc[0] = (1.0f - ax) * (1.0f - ay);
    xi[1] = (int)x0f + 1; yi[1] = (int)y0f;     wc[1] = ax * (1.0f - ay);
    xi[2] = (int)x0f;     yi[2] = (int)y0f + 1; wc[2] = (1.0f - ax) * ay;
    xi[3] = (int)x0f + 1; yi[3] = (int)y0f + 1; wc[3] = ax * ay;
    bool ok[4]; size_t pidx[4];
    size_t abase = (size_t)n * (C + 1) * hw;
    for (int k = 0; k < 4; ++k) {
        ok[k] = (xi[k] >= 0) && (xi[k] < W) && (yi[k] >= 0) && (yi[k] < H);
        pidx[k] = abase + (size_t)yi[k] * W + xi[k];
    }
    const float* vb = val + (size_t)n * C * hw + (size_t)y * W + x;
    for (int c = 0; c < C; ++c) {
        float v = vb[(size_t)c * hw];
        for (int k = 0; k < 4; ++k)
            if (ok[k]) atomicAdd(&accum[pidx[k] + (size_t)c * hw], v * wc[k]);
    }
    for (int k = 0; k < 4; ++k)
        if (ok[k]) atomicAdd(&accum[pidx[k] + (size_t)C * hw], wc[k]);
}

__global__ void splat_norm_kernel(const float* __restrict__ accum, float* __restrict__ out,
                                  int C, int HW, int N)
{
    long total = (long)N * C * HW;
    long idx = (long)blockIdx.x * blockDim.x + threadIdx.x;
    if (idx >= total) return;
    int p = idx % HW;
    int c = (idx / HW) % C;
    int n = idx / ((long)HW * C);
    float d = accum[((size_t)n * (C + 1) + C) * HW + p];
    out[idx] = accum[((size_t)n * (C + 1) + c) * HW + p] / (d + 1e-7f);
}

__global__ void corr_leaky_kernel(const float* __restrict__ f1, const float* __restrict__ f2,
                                  float* __restrict__ out, int C, int H, int W, int N)
{
    long total = (long)N * 81 * H * W;
    long idx = (long)blockIdx.x * blockDim.x + threadIdx.x;
    if (idx >= total) return;
    int x = idx % W;
    int y = (idx / W) % H;
    int k = (idx / ((long)W * H)) % 81;
    int n = idx / ((long)W * H * 81);
    int dy = k / 9 - 4, dx = k % 9 - 4;
    int sx = x + dx, sy = y + dy;
    size_t hw = (size_t)H * W;
    float s = 0.0f;
    if (sx >= 0 && sx < W && sy >= 0 && sy < H) {
        const float* a = f1 + (size_t)n * C * hw + (size_t)y * W + x;
        const float* b = f2 + (size_t)n * C * hw + (size_t)sy * W + sx;
        for (int c = 0; c < C; ++c) s += a[(size_t)c * hw] * b[(size_t)c * hw];
    }
    s *= (1.0f / (float)C);
    out[idx] = (s >= 0.0f) ? s : NEG_SLOPE * s;
}

// out4(f32) = base4(f32) + concat(df2, db2) from f16 NHWC stride-32 buffers
__global__ void add_flow_kernel(const float* __restrict__ base, const _Float16* __restrict__ df,
                                const _Float16* __restrict__ db, float* __restrict__ out,
                                int HW, int N)
{
    long total = (long)N * 4 * HW;
    long idx = (long)blockIdx.x * blockDim.x + threadIdx.x;
    if (idx >= total) return;
    int p = idx % HW;
    int c = (idx / HW) % 4;
    int n = idx / ((long)HW * 4);
    float d = (c < 2) ? (float)df[((size_t)n * HW + p) * 32 + c]
                      : (float)db[((size_t)n * HW + p) * 32 + (c - 2)];
    out[idx] = base[idx] + d;
}

// merged = wi0*m0*wi1*m1 / (m0+m1); masks are f16 NHWC stride-32, channel 0
__global__ void merge_kernel(const float* __restrict__ wi0, const float* __restrict__ wi1,
                             const _Float16* __restrict__ m0, const _Float16* __restrict__ m1,
                             float* __restrict__ out, int HW, int N)
{
    long total = (long)N * 3 * HW;
    long idx = (long)blockIdx.x * blockDim.x + threadIdx.x;
    if (idx >= total) return;
    int p = idx % HW;
    int n = idx / ((long)HW * 3);
    float a = (float)m0[((size_t)n * HW + p) * 32];
    float b = (float)m1[((size_t)n * HW + p) * 32];
    out[idx] = wi0[idx] * a * wi1[idx] * b / (a + b);
}

// ---------------------------------------------------------------------------
static inline long cdivl(long a, long b) { return (a + b - 1) / b; }
static inline int pad32(int c) { return (c + 31) & ~31; }

#define ELT(kern, total, ...) do {                                            \
    long _t = (total);                                                        \
    kern<<<dim3((unsigned)cdivl(_t, 256)), 256, 0, stream>>>(__VA_ARGS__);    \
} while (0)

extern "C" void kernel_launch(void* const* d_in, const int* in_sizes, int n_in,
                              void* d_out, int out_size, void* d_ws, size_t ws_size,
                              hipStream_t stream)
{
    (void)in_sizes; (void)n_in; (void)out_size; (void)ws_size;
    const int N = 2, H = 256, W = 448, PC = 32;

    const float* image0 = (const float*)d_in[0];
    const float* image1 = (const float*)d_in[1];
    const float* feat0[3] = { (const float*)d_in[2], (const float*)d_in[3], (const float*)d_in[4] };
    const float* feat1[3] = { (const float*)d_in[5], (const float*)d_in[6], (const float*)d_in[7] };
    const float* fwdT[3]  = { (const float*)d_in[8], (const float*)d_in[9], (const float*)d_in[10] };
    const float* bwdT[3]  = { (const float*)d_in[11], (const float*)d_in[12], (const float*)d_in[13] };
    const float* est_w[3][7];
    for (int i = 0; i < 3; ++i)
        for (int j = 0; j < 7; ++j)
            est_w[i][j] = (const float*)d_in[14 + i * 7 + j];
    const float* mask_b[5];
    const float* mask_w[5];
    for (int j = 0; j < 5; ++j) {
        mask_b[j] = (const float*)d_in[35 + j];
        mask_w[j] = (const float*)d_in[40 + j];
    }
    float* out = (float*)d_out;

    const int HW2 = H * W;
    char* wsp = (char*)d_ws;
    size_t off = 0;
    auto alloc = [&](size_t nbytes) -> void* {
        void* p = (void*)(wsp + off);
        off += (nbytes + 255) & ~(size_t)255;
        return p;
    };
    auto allocf = [&](size_t n) -> float*    { return (float*)alloc(n * 4); };
    auto alloch = [&](size_t n) -> _Float16* { return (_Float16*)alloc(n * 2); };

    // f32 NCHW buffers
    float* flow_a = allocf((size_t)4 * HW2 * N);
    float* flow_b = allocf((size_t)4 * HW2 * N);
    float* accum  = allocf((size_t)33 * HW2 * N);
    float* w0buf  = allocf((size_t)32 * HW2 * N);
    float* w1buf  = allocf((size_t)32 * HW2 * N);
    float* vol    = allocf((size_t)81 * HW2 * N);
    float* wi0    = allocf((size_t)3 * HW2 * N);
    float* wi1    = allocf((size_t)3 * HW2 * N);
    // f16 NHWC activation buffers (channel stride padded to 32)
    _Float16* cbuf  = alloch((size_t)160 * HW2 * N);   // max pad32(145)*HW2
    _Float16* ping  = alloch((size_t)64  * HW2 * N);
    _Float16* pong  = alloch((size_t)64  * HW2 * N);
    _Float16* dfbuf = alloch((size_t)32 * HW2 * N);
    _Float16* dbbuf = alloch((size_t)32 * HW2 * N);
    _Float16* m0buf = alloch((size_t)32 * HW2 * N);
    _Float16* m1buf = alloch((size_t)32 * HW2 * N);
    _Float16* zpage = alloch(128);                      // zeroed halo page

    // f16 weight arena [O][9][I]
    const int pcs[3] = { PC * 4, PC * 2, PC };
    _Float16* west[3][7];
    _Float16* wmsk[5];
    int eO[3][7], eI[3][7];
    for (int i = 0; i < 3; ++i) {
        int pc = pcs[i];
        int sh[7][2] = { {2*pc, 2*pc+81}, {2*pc, 2*pc}, {2*pc, 2*pc}, {pc, 2*pc},
                         {pc, pc}, {pc/2, pc}, {2, pc/2} };
        for (int j = 0; j < 7; ++j) {
            eO[i][j] = sh[j][0]; eI[i][j] = sh[j][1];
            west[i][j] = alloch((size_t)sh[j][0] * sh[j][1] * 9);
        }
    }
    int mO[5] = { PC, PC/2, PC/2, 4, 1 };
    int mI[5] = { 2*PC, PC, PC/2, PC/2, 4 };
    for (int j = 0; j < 5; ++j) wmsk[j] = alloch((size_t)mO[j] * mI[j] * 9);

    hipMemsetAsync(zpage, 0, 256, stream);
    for (int i = 0; i < 3; ++i)
        for (int j = 0; j < 7; ++j)
            ELT(wcvt_kernel, (long)eO[i][j] * eI[i][j] * 9, est_w[i][j], west[i][j], eO[i][j], eI[i][j]);
    for (int j = 0; j < 5; ++j)
        ELT(wcvt_kernel, (long)mO[j] * mI[j] * 9, mask_w[j], wmsk[j], mO[j], mI[j]);

    auto conv = [&](const _Float16* in, const _Float16* w, const float* b, _Float16* o,
                    int Cin, int Cins, int Cout, int Couts, int Hh, int Ww, int lk) {
        int tilesW  = (Ww + TILE_W - 1) / TILE_W;
        int tilesCo = (Cout + 15) / 16;
        dim3 grid(tilesW * tilesCo, Hh, N);
        conv3x3_wmma_kernel<<<grid, 128, 0, stream>>>(in, w, b, o, zpage,
                                                      Cin, Cins, Cout, Couts, Hh, Ww, lk);
    };

    const int Hs[3]  = { H / 4, H / 2, H };
    const int Wsz[3] = { W / 4, W / 2, W };

    hipMemsetAsync(flow_a, 0, (size_t)4 * (H / 8) * (W / 8) * N * sizeof(float), stream);

    for (int i = 0; i < 3; ++i) {
        const int pc = pcs[i], Hh = Hs[i], Ww = Wsz[i], HWc = Hh * Ww;
        const int ph = Hh / 2, pw = Ww / 2;
        const int Ccat = 2 * pc + 81, Cs = pad32(Ccat);
        const int Ch = pad32(pc / 2);

        ELT(up2x_scale_kernel, (long)N * 4 * Hh * Ww, flow_a, flow_b, 4, ph, pw, N, 2.0f);

        hipMemsetAsync(accum, 0, (size_t)(pc + 1) * HWc * N * sizeof(float), stream);
        ELT(splat_scatter_kernel, (long)N * HWc, feat0[i], flow_b, accum, pc, Hh, Ww, N, 0);
        ELT(splat_norm_kernel, (long)N * pc * HWc, accum, w0buf, pc, HWc, N);
        hipMemsetAsync(accum, 0, (size_t)(pc + 1) * HWc * N * sizeof(float), stream);
        ELT(splat_scatter_kernel, (long)N * HWc, feat1[i], flow_b, accum, pc, Hh, Ww, N, 2);
        ELT(splat_norm_kernel, (long)N * pc * HWc, accum, w1buf, pc, HWc, N);

        ELT(corr_leaky_kernel, (long)N * 81 * HWc, w0buf, w1buf, vol, pc, Hh, Ww, N);

        // forward branch
        ELT(pack_nhwc_kernel, (long)N * pc * HWc, w0buf, cbuf, pc, 0, Cs, HWc, N);
        ELT(pack_nhwc_kernel, (long)N * pc * HWc, fwdT[i], cbuf, pc, pc, Cs, HWc, N);
        ELT(pack_nhwc_kernel, (long)N * 81 * HWc, vol, cbuf, 81, 2 * pc, Cs, HWc, N);
        conv(cbuf, west[i][0], nullptr, ping, Ccat, Cs, 2*pc, 2*pc, Hh, Ww, 0);
        conv(ping, west[i][1], nullptr, pong, 2*pc, 2*pc, 2*pc, 2*pc, Hh, Ww, 0);
        conv(pong, west[i][2], nullptr, ping, 2*pc, 2*pc, 2*pc, 2*pc, Hh, Ww, 1);
        conv(ping, west[i][3], nullptr, pong, 2*pc, 2*pc, pc, pc, Hh, Ww, 0);
        conv(pong, west[i][4], nullptr, ping, pc, pc, pc, pc, Hh, Ww, 0);
        conv(ping, west[i][5], nullptr, pong, pc, pc, pc/2, Ch, Hh, Ww, 0);
        conv(pong, west[i][6], nullptr, dfbuf, pc/2, Ch, 2, 32, Hh, Ww, 1);

        // backward branch (vol slab already packed)
        ELT(pack_nhwc_kernel, (long)N * pc * HWc, w1buf, cbuf, pc, 0, Cs, HWc, N);
        ELT(pack_nhwc_kernel, (long)N * pc * HWc, bwdT[i], cbuf, pc, pc, Cs, HWc, N);
        conv(cbuf, west[i][0], nullptr, ping, Ccat, Cs, 2*pc, 2*pc, Hh, Ww, 0);
        conv(ping, west[i][1], nullptr, pong, 2*pc, 2*pc, 2*pc, 2*pc, Hh, Ww, 0);
        conv(pong, west[i][2], nullptr, ping, 2*pc, 2*pc, 2*pc, 2*pc, Hh, Ww, 1);
        conv(ping, west[i][3], nullptr, pong, 2*pc, 2*pc, pc, pc, Hh, Ww, 0);
        conv(pong, west[i][4], nullptr, ping, pc, pc, pc, pc, Hh, Ww, 0);
        conv(ping, west[i][5], nullptr, pong, pc, pc, pc/2, Ch, Hh, Ww, 0);
        conv(pong, west[i][6], nullptr, dbbuf, pc/2, Ch, 2, 32, Hh, Ww, 1);

        ELT(add_flow_kernel, (long)N * 4 * HWc, flow_b, dfbuf, dbbuf, flow_a, HWc, N);
    }

    // final warps + masks
    hipMemsetAsync(accum, 0, (size_t)33 * HW2 * N * sizeof(float), stream);
    ELT(splat_scatter_kernel, (long)N * HW2, feat0[2], flow_a, accum, PC, H, W, N, 0);
    ELT(splat_norm_kernel, (long)N * PC * HW2, accum, w0buf, PC, HW2, N);
    hipMemsetAsync(accum, 0, (size_t)33 * HW2 * N * sizeof(float), stream);
    ELT(splat_scatter_kernel, (long)N * HW2, feat1[2], flow_a, accum, PC, H, W, N, 2);
    ELT(splat_norm_kernel, (long)N * PC * HW2, accum, w1buf, PC, HW2, N);

    ELT(pack_nhwc_kernel, (long)N * PC * HW2, w0buf, cbuf, PC, 0, 64, HW2, N);
    ELT(pack_nhwc_kernel, (long)N * PC * HW2, feat0[2], cbuf, PC, PC, 64, HW2, N);
    conv(cbuf, wmsk[0], mask_b[0], ping, 2*PC, 64, PC, 32, H, W, 0);
    conv(ping, wmsk[1], mask_b[1], pong, PC, 32, PC/2, 32, H, W, 0);
    conv(pong, wmsk[2], mask_b[2], ping, PC/2, 32, PC/2, 32, H, W, 0);
    conv(ping, wmsk[3], mask_b[3], pong, PC/2, 32, 4, 32, H, W, 0);
    conv(pong, wmsk[4], mask_b[4], m0buf, 4, 32, 1, 32, H, W, 1);

    ELT(pack_nhwc_kernel, (long)N * PC * HW2, w1buf, cbuf, PC, 0, 64, HW2, N);
    ELT(pack_nhwc_kernel, (long)N * PC * HW2, feat1[2], cbuf, PC, PC, 64, HW2, N);
    conv(cbuf, wmsk[0], mask_b[0], ping, 2*PC, 64, PC, 32, H, W, 0);
    conv(ping, wmsk[1], mask_b[1], pong, PC, 32, PC/2, 32, H, W, 0);
    conv(pong, wmsk[2], mask_b[2], ping, PC/2, 32, PC/2, 32, H, W, 0);
    conv(ping, wmsk[3], mask_b[3], pong, PC/2, 32, 4, 32, H, W, 0);
    conv(pong, wmsk[4], mask_b[4], m1buf, 4, 32, 1, 32, H, W, 1);

    hipMemsetAsync(accum, 0, (size_t)4 * HW2 * N * sizeof(float), stream);
    ELT(splat_scatter_kernel, (long)N * HW2, image0, flow_a, accum, 3, H, W, N, 0);
    ELT(splat_norm_kernel, (long)N * 3 * HW2, accum, wi0, 3, HW2, N);
    hipMemsetAsync(accum, 0, (size_t)4 * HW2 * N * sizeof(float), stream);
    ELT(splat_scatter_kernel, (long)N * HW2, image1, flow_a, accum, 3, H, W, N, 2);
    ELT(splat_norm_kernel, (long)N * 3 * HW2, accum, wi1, 3, HW2, N);

    ELT(merge_kernel, (long)N * 3 * HW2, wi0, wi1, m0buf, m1buf, out, HW2, N);
}
